// hypercorre_topk2_82008105550150
// MI455X (gfx1250) — matrix operand order repaired and checked by reading, not compiled, not run on hardware
//
#include <hip/hip_runtime.h>
#include <math.h>

// ---------------------------------------------------------------------------
// Fixed problem sizes from setup_inputs(): B=4, T=4 (t=3), N=4096 (=64x64),
// C=256, K=100, 4C=1024, 8 heads of 32.
// ---------------------------------------------------------------------------
#define B_   4
#define T_   4
#define N_   4096
#define C_   256
#define K_   100
#define C4_  1024
#define NH_  8
#define HD_  32
#define HP_  70   // padded image height (64 + 2*3)
#define WP_  72   // padded image width  (64 + 2*3, +2 for alignment)

typedef __attribute__((ext_vector_type(8)))  float  v8f;
typedef __attribute__((ext_vector_type(8)))  __bf16 v8bf;
typedef __attribute__((ext_vector_type(16))) __bf16 v16bf;

#define WMMA_BF16(a, b, c) \
  __builtin_amdgcn_wmma_f32_16x16x32_bf16(false, (a), false, (b), (short)0, (c), false, false)

__device__ __forceinline__ float gelu_f(float x) {
  return 0.5f * x * (1.0f + erff(x * 0.7071067811865476f));
}

__device__ __forceinline__ v8f vzero8() {
  v8f v;
#pragma unroll
  for (int i = 0; i < 8; ++i) v[i] = 0.0f;
  return v;
}

// A fragment (16x32 bf16): lane covers row m = lane&15.
// Half-wave 0 holds K {0..7, 16..23}, half-wave 1 holds K {8..15, 24..31}.
// p must already include row*ld + k0 + ((lane>>4)<<3).
__device__ __forceinline__ v16bf frag_a(const __bf16* p) {
  v8bf lo = *(const v8bf*)(p);
  v8bf hi = *(const v8bf*)(p + 16);
  return __builtin_shufflevector(lo, hi, 0, 1, 2, 3, 4, 5, 6, 7, 8, 9, 10, 11, 12, 13, 14, 15);
}

// B fragment (32x16 bf16, stored as B^T i.e. N x K row-major): lane covers
// column n = lane&15; half-wave 0 holds K {0..7, 16..23}, half-wave 1 holds K {8..15, 24..31}.
// p must already include n*ld + k0 + ((lane>>4)<<3).
__device__ __forceinline__ v16bf frag_b(const __bf16* p) {
  v8bf lo = *(const v8bf*)(p);
  v8bf hi = *(const v8bf*)(p + 16);
  return __builtin_shufflevector(lo, hi, 0, 1, 2, 3, 4, 5, 6, 7, 8, 9, 10, 11, 12, 13, 14, 15);
}

// D fragment store helper: lane holds column col, rows rowBase..rowBase+7.
template <bool GUARD>
__device__ __forceinline__ void store_acc(v8f a, int rowBase, int col,
                                          const float* bias, float* outF, __bf16* outB,
                                          long long obase, int M, int Nd, int doGelu) {
  if (GUARD && col >= Nd) return;
  float bv = bias ? bias[col] : 0.0f;
#pragma unroll
  for (int r = 0; r < 8; ++r) {
    int row = rowBase + r;
    if (!GUARD || row < M) {
      float val = a[r] + bv;
      if (doGelu) val = gelu_f(val);
      long long o = obase + (long long)row * Nd + col;
      if (outF) outF[o] = val;
      if (outB) outB[o] = (__bf16)val;
    }
  }
}

// ---------------------------------------------------------------------------
// Generic batched GEMM:  out[bat] = A[bat] (MxK) @ B[bat]^T (NdxK) + bias
// block = 128 threads = 4 waves (2x2); each wave owns a 32x64 tile
// (2 A-frags x 4 B-frags -> 8 WMMA per 32-K step); block tile 64x128.
// GUARD=false requires M%64==0 and Nd%128==0. For GUARD=true, out-of-range
// rows are clamped for addressing (loads stay in-bounds) and discarded at the
// store, so the inner loop carries no predication at all.
// ---------------------------------------------------------------------------
template <bool GUARD>
__global__ void __launch_bounds__(128) gemm_bf16_kernel(
    const __bf16* __restrict__ A, const __bf16* __restrict__ B,
    const float* __restrict__ bias, float* __restrict__ outF, __bf16* __restrict__ outB,
    int M, int Nd, int Kd, long long sA, long long sB, long long sO, int doGelu) {
  const int tid = threadIdx.x;
  const int lane = tid & 31;
  const int wave = tid >> 5;
  const int wm = wave >> 1, wn = wave & 1;
  const int bat = blockIdx.z;
  const int tm = blockIdx.x * 64, tn = blockIdx.y * 128;
  const int l16 = lane & 15;
  const int ka = (lane >> 4) << 3;   // A fragment K sub-offset
  const int kb = (lane >> 4) << 3;   // B fragment K sub-offset

  A += (long long)bat * sA;
  B += (long long)bat * sB;

  const int rowA0 = tm + wm * 32 + l16;
  const int rowA1 = rowA0 + 16;
  const int colB0 = tn + wn * 64 + l16;
  const int colB1 = colB0 + 16, colB2 = colB0 + 32, colB3 = colB0 + 48;

  // clamped addressing rows (results for clamped lanes are never stored)
  int rA0 = rowA0, rA1 = rowA1, rB0 = colB0, rB1 = colB1, rB2 = colB2, rB3 = colB3;
  if (GUARD) {
    const int mc = M - 1, nc = Nd - 1;
    rA0 = rA0 < mc ? rA0 : mc;
    rA1 = rA1 < mc ? rA1 : mc;
    rB0 = rB0 < nc ? rB0 : nc;
    rB1 = rB1 < nc ? rB1 : nc;
    rB2 = rB2 < nc ? rB2 : nc;
    rB3 = rB3 < nc ? rB3 : nc;
  }

  const __bf16* pa0 = A + (long long)rA0 * Kd + ka;
  const __bf16* pa1 = A + (long long)rA1 * Kd + ka;
  const __bf16* pb0 = B + (long long)rB0 * Kd + kb;
  const __bf16* pb1 = B + (long long)rB1 * Kd + kb;
  const __bf16* pb2 = B + (long long)rB2 * Kd + kb;
  const __bf16* pb3 = B + (long long)rB3 * Kd + kb;

  v8f acc00 = vzero8(), acc01 = vzero8(), acc02 = vzero8(), acc03 = vzero8();
  v8f acc10 = vzero8(), acc11 = vzero8(), acc12 = vzero8(), acc13 = vzero8();

  for (int k0 = 0; k0 < Kd; k0 += 32) {
    v16bf a0 = frag_a(pa0 + k0);
    v16bf a1 = frag_a(pa1 + k0);
    v16bf b0 = frag_b(pb0 + k0);
    v16bf b1 = frag_b(pb1 + k0);
    v16bf b2 = frag_b(pb2 + k0);
    v16bf b3 = frag_b(pb3 + k0);
    acc00 = WMMA_BF16(a0, b0, acc00);
    acc01 = WMMA_BF16(a0, b1, acc01);
    acc02 = WMMA_BF16(a0, b2, acc02);
    acc03 = WMMA_BF16(a0, b3, acc03);
    acc10 = WMMA_BF16(a1, b0, acc10);
    acc11 = WMMA_BF16(a1, b1, acc11);
    acc12 = WMMA_BF16(a1, b2, acc12);
    acc13 = WMMA_BF16(a1, b3, acc13);
  }

  const long long obase = (long long)bat * sO;
  const int row0 = tm + wm * 32 + ((lane >> 4) << 3);
  const int row1 = row0 + 16;
  store_acc<GUARD>(acc00, row0, colB0, bias, outF, outB, obase, M, Nd, doGelu);
  store_acc<GUARD>(acc01, row0, colB1, bias, outF, outB, obase, M, Nd, doGelu);
  store_acc<GUARD>(acc02, row0, colB2, bias, outF, outB, obase, M, Nd, doGelu);
  store_acc<GUARD>(acc03, row0, colB3, bias, outF, outB, obase, M, Nd, doGelu);
  store_acc<GUARD>(acc10, row1, colB0, bias, outF, outB, obase, M, Nd, doGelu);
  store_acc<GUARD>(acc11, row1, colB1, bias, outF, outB, obase, M, Nd, doGelu);
  store_acc<GUARD>(acc12, row1, colB2, bias, outF, outB, obase, M, Nd, doGelu);
  store_acc<GUARD>(acc13, row1, colB3, bias, outF, outB, obase, M, Nd, doGelu);
}

// ---------------------------------------------------------------------------
// conv1 7x7 pad 3 as implicit GEMM over 49 shifted 1x1 convs, NHWC layout,
// reading a zero-padded image xp (bt, 70, 72, 256) so the hot loop is fully
// branch-free. B = w1r[tap] (256 x 256, (co,ci) row-major). Tiling identical
// to gemm_bf16_kernel (wave = 32x64). Epilogue: +bias, GELU, bf16 store.
// ---------------------------------------------------------------------------
__global__ void __launch_bounds__(128) conv1_kernel(
    const __bf16* __restrict__ xp, const __bf16* __restrict__ w1r,
    const float* __restrict__ bias, __bf16* __restrict__ out) {
  const int tid = threadIdx.x;
  const int lane = tid & 31;
  const int wave = tid >> 5;
  const int wm = wave >> 1, wn = wave & 1;
  const int bt = blockIdx.z;
  const int tm = blockIdx.x * 64, tn = blockIdx.y * 128;
  const int l16 = lane & 15;
  const int ka = (lane >> 4) << 3;
  const int kb = (lane >> 4) << 3;

  const int p0 = tm + wm * 32 + l16;  // output pixel for A frag 0
  const int p1 = p0 + 16;             // output pixel for A frag 1
  const int h0 = p0 >> 6, w0 = p0 & 63;
  const int h1 = p1 >> 6, w1 = p1 & 63;
  const int colB0 = tn + wn * 64 + l16;
  const int colB1 = colB0 + 16, colB2 = colB0 + 32, colB3 = colB0 + 48;

  // padded-image base pointers for tap (0,0); taps shift by (dh*WP_+dw)*C_
  const __bf16* a0base = xp + (((long long)bt * HP_ + (h0 + 3)) * WP_ + (w0 + 3)) * C_ + ka;
  const __bf16* a1base = xp + (((long long)bt * HP_ + (h1 + 3)) * WP_ + (w1 + 3)) * C_ + ka;

  v8f acc00 = vzero8(), acc01 = vzero8(), acc02 = vzero8(), acc03 = vzero8();
  v8f acc10 = vzero8(), acc11 = vzero8(), acc12 = vzero8(), acc13 = vzero8();

#pragma unroll 1
  for (int dh = -3; dh <= 3; ++dh) {
#pragma unroll 1
    for (int dw = -3; dw <= 3; ++dw) {
      const int tap = (dh + 3) * 7 + (dw + 3);
      const long long shift = (long long)(dh * WP_ + dw) * C_;
      const __bf16* pa0 = a0base + shift;
      const __bf16* pa1 = a1base + shift;
      const __bf16* pb0 = w1r + ((long long)tap * C_ + colB0) * C_ + kb;
      const __bf16* pb1 = w1r + ((long long)tap * C_ + colB1) * C_ + kb;
      const __bf16* pb2 = w1r + ((long long)tap * C_ + colB2) * C_ + kb;
      const __bf16* pb3 = w1r + ((long long)tap * C_ + colB3) * C_ + kb;
#pragma unroll
      for (int k0 = 0; k0 < C_; k0 += 32) {
        v16bf a0 = frag_a(pa0 + k0);
        v16bf a1 = frag_a(pa1 + k0);
        v16bf b0 = frag_b(pb0 + k0);
        v16bf b1 = frag_b(pb1 + k0);
        v16bf b2 = frag_b(pb2 + k0);
        v16bf b3 = frag_b(pb3 + k0);
        acc00 = WMMA_BF16(a0, b0, acc00);
        acc01 = WMMA_BF16(a0, b1, acc01);
        acc02 = WMMA_BF16(a0, b2, acc02);
        acc03 = WMMA_BF16(a0, b3, acc03);
        acc10 = WMMA_BF16(a1, b0, acc10);
        acc11 = WMMA_BF16(a1, b1, acc11);
        acc12 = WMMA_BF16(a1, b2, acc12);
        acc13 = WMMA_BF16(a1, b3, acc13);
      }
    }
  }

  const long long obase = (long long)bt * N_ * C_;
  const int row0 = tm + wm * 32 + ((lane >> 4) << 3);
  const int row1 = row0 + 16;
  store_acc<false>(acc00, row0, colB0, bias, nullptr, out, obase, N_, C_, 1);
  store_acc<false>(acc01, row0, colB1, bias, nullptr, out, obase, N_, C_, 1);
  store_acc<false>(acc02, row0, colB2, bias, nullptr, out, obase, N_, C_, 1);
  store_acc<false>(acc03, row0, colB3, bias, nullptr, out, obase, N_, C_, 1);
  store_acc<false>(acc10, row1, colB0, bias, nullptr, out, obase, N_, C_, 1);
  store_acc<false>(acc11, row1, colB1, bias, nullptr, out, obase, N_, C_, 1);
  store_acc<false>(acc12, row1, colB2, bias, nullptr, out, obase, N_, C_, 1);
  store_acc<false>(acc13, row1, colB3, bias, nullptr, out, obase, N_, C_, 1);
}

// ---------------------------------------------------------------------------
// Block reductions (blockDim.x == 256)
// ---------------------------------------------------------------------------
__device__ __forceinline__ float brs_sum(float v, float* sm) {
  const int t = threadIdx.x;
  sm[t] = v;
  __syncthreads();
#pragma unroll
  for (int s = 128; s > 0; s >>= 1) {
    if (t < s) sm[t] += sm[t + s];
    __syncthreads();
  }
  float r = sm[0];
  __syncthreads();
  return r;
}

__device__ __forceinline__ float brs_max(float v, float* sm) {
  const int t = threadIdx.x;
  sm[t] = v;
  __syncthreads();
#pragma unroll
  for (int s = 128; s > 0; s >>= 1) {
    if (t < s) sm[t] = fmaxf(sm[t], sm[t + s]);
    __syncthreads();
  }
  float r = sm[0];
  __syncthreads();
  return r;
}

// ---------------------------------------------------------------------------
// Packing kernels
// ---------------------------------------------------------------------------
__global__ void pack_xt_kernel(const float* __restrict__ x, const float* __restrict__ mem,
                               __bf16* __restrict__ xt, __bf16* __restrict__ xtT) {
  long long idx = (long long)blockIdx.x * 256 + threadIdx.x;  // 16*4096*256 exact
  int c = (int)(idx & 255);
  long long r = idx >> 8;
  int n = (int)(r & 4095);
  int bt = (int)(r >> 12);
  int b = bt >> 2, s = bt & 3;
  float val = (s < 3) ? mem[(((long long)b * 3 + s) * N_ + n) * C_ + c]
                      : x[((long long)b * N_ + n) * C_ + c];
  xt[idx] = (__bf16)val;
  xtT[((long long)bt * C_ + c) * (long long)N_ + n] = (__bf16)val;
}

// zero-padded NHWC image for conv1: xp (bt, 70, 72, 256) bf16
__global__ void pad_xt_kernel(const float* __restrict__ x, const float* __restrict__ mem,
                              __bf16* __restrict__ xp) {
  long long idx = (long long)blockIdx.x * 256 + threadIdx.x;  // 16*70*72*256 exact
  int c = (int)(idx & 255);
  long long r = idx >> 8;  // bt*70*72 + hp*72 + wp
  int wp = (int)(r % WP_);
  r /= WP_;
  int hp = (int)(r % HP_);
  int bt = (int)(r / HP_);
  float val = 0.0f;
  const int h = hp - 3, w = wp - 3;
  if ((unsigned)h < 64u && (unsigned)w < 64u) {
    const int n = (h << 6) + w;
    const int b = bt >> 2, s = bt & 3;
    val = (s < 3) ? mem[(((long long)b * 3 + s) * N_ + n) * C_ + c]
                  : x[((long long)b * N_ + n) * C_ + c];
  }
  xp[idx] = (__bf16)val;
}

__global__ void pack_w1_kernel(const float* __restrict__ w, __bf16* __restrict__ w1r) {
  long long i = (long long)blockIdx.x * 256 + threadIdx.x;  // 256*256*49
  if (i >= 256LL * 256 * 49) return;
  int tap = (int)(i % 49);
  long long r = i / 49;
  int ci = (int)(r & 255);
  int co = (int)(r >> 8);
  w1r[((long long)tap * C_ + co) * C_ + ci] = (__bf16)w[i];
}

__global__ void f32_to_bf16_kernel(const float* __restrict__ in, __bf16* __restrict__ out,
                                   long long total) {
  long long i = (long long)blockIdx.x * 256 + threadIdx.x;
  if (i < total) out[i] = (__bf16)in[i];
}

// ---------------------------------------------------------------------------
// softmax over N per (bt,k): cz is (bt, n, 100) f32 -> probs (bt, k, n) bf16
// ---------------------------------------------------------------------------
__global__ void __launch_bounds__(256) softmax_kernel(const float* __restrict__ cz,
                                                      __bf16* __restrict__ probs) {
  __shared__ float sm[256];
  const int k = blockIdx.x % K_;
  const int bt = blockIdx.x / K_;
  const int tid = threadIdx.x;
  const float* src = cz + (long long)bt * N_ * K_ + k;
  float vals[16];
  float mx = -3.4e38f;
#pragma unroll
  for (int i = 0; i < 16; ++i) {
    int n = i * 256 + tid;
    float v = src[(long long)n * K_];
    vals[i] = v;
    mx = fmaxf(mx, v);
  }
  mx = brs_max(mx, sm);
  float s = 0.0f;
#pragma unroll
  for (int i = 0; i < 16; ++i) {
    vals[i] = expf(vals[i] - mx);
    s += vals[i];
  }
  s = brs_sum(s, sm);
  const float inv = 1.0f / s;
  __bf16* dst = probs + ((long long)bt * K_ + k) * (long long)N_;
#pragma unroll
  for (int i = 0; i < 16; ++i) dst[i * 256 + tid] = (__bf16)(vals[i] * inv);
}

// ---------------------------------------------------------------------------
// cosine-sim combine of centers + LayerNorm -> C_in (bf16), one block per (b,k)
// center: (B, T, K, C) f32
// ---------------------------------------------------------------------------
__global__ void __launch_bounds__(256) combine_kernel(
    const float* __restrict__ center, const float* __restrict__ alpha_p,
    const float* __restrict__ beta_p, const float* __restrict__ lnw,
    const float* __restrict__ lnb, __bf16* __restrict__ cinb) {
  __shared__ float sm[256];
  const int b = blockIdx.x / K_;
  const int k = blockIdx.x % K_;
  const int c = threadIdx.x;
  const long long stride_s = (long long)K_ * C_;
  const float* base = center + ((long long)b * T_ * K_ + k) * (long long)C_ + c;
  const float p0 = base[0 * stride_s];
  const float p1 = base[1 * stride_s];
  const float p2 = base[2 * stride_s];
  const float last = base[3 * stride_s];

  const float d0 = brs_sum(last * p0, sm);
  const float d1 = brs_sum(last * p1, sm);
  const float d2 = brs_sum(last * p2, sm);
  const float nl = sqrtf(brs_sum(last * last, sm));
  const float n0 = sqrtf(brs_sum(p0 * p0, sm));
  const float n1 = sqrtf(brs_sum(p1 * p1, sm));
  const float n2 = sqrtf(brs_sum(p2 * p2, sm));

  const float alpha = alpha_p[0], beta = beta_p[0];
  const float c0 = 1.0f / (1.0f + expf(-(beta + alpha * (d0 / fmaxf(nl * n0, 1e-8f)))));
  const float c1 = 1.0f / (1.0f + expf(-(beta + alpha * (d1 / fmaxf(nl * n1, 1e-8f)))));
  const float c2 = 1.0f / (1.0f + expf(-(beta + alpha * (d2 / fmaxf(nl * n2, 1e-8f)))));

  const float cin = last + c0 * p0 + c1 * p1 + c2 * p2;
  const float m = brs_sum(cin, sm) * (1.0f / C_);
  const float var = brs_sum((cin - m) * (cin - m), sm) * (1.0f / C_);
  const float y = (cin - m) * rsqrtf(var + 1e-5f) * lnw[c] + lnb[c];
  cinb[((long long)b * K_ + k) * (long long)C_ + c] = (__bf16)y;
}

// ---------------------------------------------------------------------------
// out = res + LN(inp), row-wise over C=256; one block per row
// ---------------------------------------------------------------------------
__global__ void __launch_bounds__(256) ln_add_kernel(
    const float* __restrict__ inp, const float* __restrict__ res,
    const float* __restrict__ lnw, const float* __restrict__ lnb,
    float* __restrict__ outF, __bf16* __restrict__ outB) {
  __shared__ float sm[256];
  const long long row = blockIdx.x;
  const int c = threadIdx.x;
  const float xv = inp[row * C_ + c];
  const float m = brs_sum(xv, sm) * (1.0f / C_);
  const float var = brs_sum((xv - m) * (xv - m), sm) * (1.0f / C_);
  const float y = res[row * C_ + c] + ((xv - m) * rsqrtf(var + 1e-5f) * lnw[c] + lnb[c]);
  if (outF) outF[row * C_ + c] = y;
  if (outB) outB[row * C_ + c] = (__bf16)y;
}

// ---------------------------------------------------------------------------
// attention core: K=100 keys, hd=32; online softmax; one thread per query row
// ---------------------------------------------------------------------------
__global__ void __launch_bounds__(256) attn_kernel(
    const float* __restrict__ q, const float* __restrict__ kk,
    const float* __restrict__ vv, __bf16* __restrict__ o) {
  __shared__ float ks[K_ * HD_];
  __shared__ float vs[K_ * HD_];
  const int b = blockIdx.z, h = blockIdx.y;
  const int n = blockIdx.x * 256 + threadIdx.x;
  for (int idx = threadIdx.x; idx < K_ * HD_; idx += 256) {
    const int j = idx / HD_, d = idx % HD_;
    ks[idx] = kk[((long long)b * K_ + j) * C_ + h * HD_ + d];
    vs[idx] = vv[((long long)b * K_ + j) * C_ + h * HD_ + d];
  }
  __syncthreads();

  float qr[HD_];
  const float* qp = q + ((long long)b * N_ + n) * C_ + h * HD_;
#pragma unroll
  for (int d = 0; d < HD_; ++d) qr[d] = qp[d] * 0.17677669529663687f;  // hd^-0.5

  float m = -3.4e38f, l = 0.0f;
  float acc[HD_];
#pragma unroll
  for (int d = 0; d < HD_; ++d) acc[d] = 0.0f;

  for (int j = 0; j < K_; ++j) {
    const float* kj = &ks[j * HD_];
    float s = 0.0f;
#pragma unroll
    for (int d = 0; d < HD_; ++d) s += qr[d] * kj[d];
    const float mn = fmaxf(m, s);
    const float corr = expf(m - mn);
    const float p = expf(s - mn);
    l = l * corr + p;
    const float* vj = &vs[j * HD_];
#pragma unroll
    for (int d = 0; d < HD_; ++d) acc[d] = acc[d] * corr + p * vj[d];
    m = mn;
  }
  const float inv = 1.0f / l;
  __bf16* op = o + ((long long)b * N_ + n) * C_ + h * HD_;
#pragma unroll
  for (int d = 0; d < HD_; ++d) op[d] = (__bf16)(acc[d] * inv);
}

// ---------------------------------------------------------------------------
// depthwise 3x3 on (B, N, 1024) bf16 + bias + GELU -> bf16
// ---------------------------------------------------------------------------
__global__ void __launch_bounds__(256) dwconv_kernel(
    const __bf16* __restrict__ y1, const float* __restrict__ dww,
    const float* __restrict__ dwb, __bf16* __restrict__ ydw) {
  const int ch = blockIdx.y * 256 + threadIdx.x;
  const int bn = blockIdx.x;
  const int b = bn >> 12;
  const int n = bn & 4095;
  const int hh = n >> 6, ww = n & 63;
  float acc = dwb[ch];
#pragma unroll
  for (int tap = 0; tap < 9; ++tap) {
    const int dy = tap / 3 - 1, dx = tap % 3 - 1;
    const int h2 = hh + dy, w2 = ww + dx;
    if ((unsigned)h2 < 64u && (unsigned)w2 < 64u) {
      const int n2 = (h2 << 6) + w2;
      acc += (float)y1[((long long)b * N_ + n2) * C4_ + ch] * dww[ch * 9 + tap];
    }
  }
  ydw[((long long)b * N_ + n) * C4_ + ch] = (__bf16)gelu_f(acc);
}

// ---------------------------------------------------------------------------
// host orchestration
// ---------------------------------------------------------------------------
extern "C" void kernel_launch(void* const* d_in, const int* in_sizes, int n_in,
                              void* d_out, int out_size, void* d_ws, size_t ws_size,
                              hipStream_t stream) {
  (void)in_sizes; (void)n_in; (void)out_size; (void)ws_size;

  const float* x       = (const float*)d_in[0];
  const float* mem     = (const float*)d_in[1];
  const float* conv1_w = (const float*)d_in[2];
  const float* conv1_b = (const float*)d_in[3];
  const float* conv2_w = (const float*)d_in[4];
  const float* conv2_b = (const float*)d_in[5];
  const float* conv3_w = (const float*)d_in[6];
  const float* sim_a   = (const float*)d_in[7];
  const float* sim_b   = (const float*)d_in[8];
  const float* ln_w    = (const float*)d_in[9];
  const float* ln_b    = (const float*)d_in[10];
  const float* q_w     = (const float*)d_in[11];
  const float* q_b     = (const float*)d_in[12];
  const float* k_w     = (const float*)d_in[13];
  const float* k_b     = (const float*)d_in[14];
  const float* v_w     = (const float*)d_in[15];
  const float* v_b     = (const float*)d_in[16];
  const float* proj_w  = (const float*)d_in[17];
  const float* proj_b  = (const float*)d_in[18];
  const float* fc1_w   = (const float*)d_in[19];
  const float* fc1_b   = (const float*)d_in[20];
  const float* dw_w    = (const float*)d_in[21];
  const float* dw_b    = (const float*)d_in[22];
  const float* fc2_w   = (const float*)d_in[23];
  const float* fc2_b   = (const float*)d_in[24];
  // d_in[25..27] = H, W, t (fixed at 64, 64, 3)

  char* ws = (char*)d_ws;
  size_t off = 0;
  auto alloc = [&](size_t bytes) -> char* {
    char* p = ws + off;
    off += (bytes + 255) & ~(size_t)255;
    return p;
  };

  const long long BT = 16;  // B*T
  __bf16* xt_bf   = (__bf16*)alloc(BT * N_ * C_ * 2);          // (bt, n, c)
  __bf16* xtT_bf  = (__bf16*)alloc(BT * C_ * N_ * 2);          // (bt, c, n)
  __bf16* xp_bf   = (__bf16*)alloc(BT * HP_ * WP_ * C_ * 2);   // padded NHWC image
  __bf16* w1r     = (__bf16*)alloc(49LL * C_ * C_ * 2);        // (tap, co, ci)
  __bf16* w2b     = (__bf16*)alloc((long long)C_ * C_ * 2);
  __bf16* w3b     = (__bf16*)alloc((long long)K_ * C_ * 2);
  __bf16* qwb     = (__bf16*)alloc((long long)C_ * C_ * 2);
  __bf16* kwb     = (__bf16*)alloc((long long)C_ * C_ * 2);
  __bf16* vwb     = (__bf16*)alloc((long long)C_ * C_ * 2);
  __bf16* pwb     = (__bf16*)alloc((long long)C_ * C_ * 2);
  __bf16* fc1wb   = (__bf16*)alloc((long long)C4_ * C_ * 2);
  __bf16* fc2wb   = (__bf16*)alloc((long long)C_ * C4_ * 2);
  __bf16* h1      = (__bf16*)alloc(BT * N_ * C_ * 2);          // conv1 out (reused as y1)
  __bf16* h2      = (__bf16*)alloc(BT * N_ * C_ * 2);          // conv2 out (reused as ydw)
  float*  cz      = (float*)alloc(BT * N_ * K_ * 4);           // conv3 out (reused as y2)
  __bf16* probs   = (__bf16*)alloc(BT * K_ * N_ * 2);          // softmax (bt, k, n)
  float*  centerf = (float*)alloc(BT * K_ * C_ * 4);           // (b, t, k, c)
  __bf16* cinb    = (__bf16*)alloc((long long)B_ * K_ * C_ * 2);
  float*  qf      = (float*)alloc((long long)B_ * N_ * C_ * 4);  // reused as oproj
  float*  kf      = (float*)alloc((long long)B_ * K_ * C_ * 4);
  float*  vf      = (float*)alloc((long long)B_ * K_ * C_ * 4);
  __bf16* o_bf    = (__bf16*)alloc((long long)B_ * N_ * C_ * 2);
  float*  outf    = (float*)alloc((long long)B_ * N_ * C_ * 4);
  __bf16* out_bf  = (__bf16*)alloc((long long)B_ * N_ * C_ * 2);
  __bf16* y1      = h1;   // alias: h1 dead after conv2
  __bf16* ydw     = h2;   // alias: h2 dead after conv3
  float*  y2      = cz;   // alias: cz dead after softmax
  float*  oproj   = qf;   // alias: qf dead after attention

  // ---- pack activations and weights to bf16 ----
  pack_xt_kernel<<<65536, 256, 0, stream>>>(x, mem, xt_bf, xtT_bf);
  pad_xt_kernel<<<80640, 256, 0, stream>>>(x, mem, xp_bf);  // 16*70*72*256 / 256
  pack_w1_kernel<<<(3211264 + 255) / 256, 256, 0, stream>>>(conv1_w, w1r);
  f32_to_bf16_kernel<<<(65536 + 255) / 256, 256, 0, stream>>>(conv2_w, w2b, 65536);
  f32_to_bf16_kernel<<<(25600 + 255) / 256, 256, 0, stream>>>(conv3_w, w3b, 25600);
  f32_to_bf16_kernel<<<(65536 + 255) / 256, 256, 0, stream>>>(q_w, qwb, 65536);
  f32_to_bf16_kernel<<<(65536 + 255) / 256, 256, 0, stream>>>(k_w, kwb, 65536);
  f32_to_bf16_kernel<<<(65536 + 255) / 256, 256, 0, stream>>>(v_w, vwb, 65536);
  f32_to_bf16_kernel<<<(65536 + 255) / 256, 256, 0, stream>>>(proj_w, pwb, 65536);
  f32_to_bf16_kernel<<<(262144 + 255) / 256, 256, 0, stream>>>(fc1_w, fc1wb, 262144);
  f32_to_bf16_kernel<<<(262144 + 255) / 256, 256, 0, stream>>>(fc2_w, fc2wb, 262144);

  // ---- conv tower ----
  conv1_kernel<<<dim3(64, 2, 16), 128, 0, stream>>>(xp_bf, w1r, conv1_b, h1);
  gemm_bf16_kernel<false><<<dim3(1024, 2, 1), 128, 0, stream>>>(   // conv2 + GELU
      h1, w2b, conv2_b, nullptr, h2, 65536, C_, C_, 0, 0, 0, 1);
  gemm_bf16_kernel<true><<<dim3(1024, 1, 1), 128, 0, stream>>>(    // conv3 (Nd=100)
      h2, w3b, nullptr, cz, nullptr, 65536, K_, C_, 0, 0, 0, 0);

  // ---- region centers ----
  softmax_kernel<<<1600, 256, 0, stream>>>(cz, probs);
  gemm_bf16_kernel<true><<<dim3(2, 2, 16), 128, 0, stream>>>(      // center einsum (M=100)
      probs, xtT_bf, nullptr, centerf, nullptr, K_, C_, N_,
      (long long)K_ * N_, (long long)C_ * N_, (long long)K_ * C_, 0);
  combine_kernel<<<400, 256, 0, stream>>>(centerf, sim_a, sim_b, ln_w, ln_b, cinb);

  // ---- attention ----
  gemm_bf16_kernel<false><<<dim3(64, 2, 4), 128, 0, stream>>>(     // q = x @ qw^T + qb
      xt_bf + 3LL * N_ * C_, qwb, q_b, qf, nullptr, N_, C_, C_,
      4LL * N_ * C_, 0, (long long)N_ * C_, 0);
  gemm_bf16_kernel<true><<<dim3(2, 2, 4), 128, 0, stream>>>(       // k (M=100)
      cinb, kwb, k_b, kf, nullptr, K_, C_, C_,
      (long long)K_ * C_, 0, (long long)K_ * C_, 0);
  gemm_bf16_kernel<true><<<dim3(2, 2, 4), 128, 0, stream>>>(       // v (M=100)
      cinb, vwb, v_b, vf, nullptr, K_, C_, C_,
      (long long)K_ * C_, 0, (long long)K_ * C_, 0);
  attn_kernel<<<dim3(16, 8, 4), 256, 0, stream>>>(qf, kf, vf, o_bf);
  gemm_bf16_kernel<false><<<dim3(256, 2, 1), 128, 0, stream>>>(    // proj
      o_bf, pwb, proj_b, oproj, nullptr, 16384, C_, C_, 0, 0, 0, 0);
  ln_add_kernel<<<16384, 256, 0, stream>>>(oproj, x, ln_w, ln_b, outf, out_bf);

  // ---- MLP with depthwise conv ----
  gemm_bf16_kernel<false><<<dim3(256, 8, 1), 128, 0, stream>>>(    // fc1
      out_bf, fc1wb, fc1_b, nullptr, y1, 16384, C4_, C_, 0, 0, 0, 0);
  dwconv_kernel<<<dim3(16384, 4), 256, 0, stream>>>(y1, dw_w, dw_b, ydw);
  gemm_bf16_kernel<false><<<dim3(256, 2, 1), 128, 0, stream>>>(    // fc2
      ydw, fc2wb, fc2_b, y2, nullptr, 16384, C_, C4_, 0, 0, 0, 0);
  ln_add_kernel<<<16384, 256, 0, stream>>>(y2, outf, ln_w, ln_b, (float*)d_out, nullptr);
}
